// ALayer_DR1_v1_light_v1_14637248545087
// MI455X (gfx1250) — matrix operand order repaired and checked by reading, not compiled, not run on hardware
//
#include <hip/hip_runtime.h>
#include <cstddef>

// ---------------------------------------------------------------------------
// Types for CDNA5 WMMA
// ---------------------------------------------------------------------------
typedef __attribute__((ext_vector_type(16))) __bf16 v16bf;
typedef __attribute__((ext_vector_type(8)))  __bf16 v8bf;
typedef __attribute__((ext_vector_type(8)))  float  v8f;

__device__ __forceinline__ float sigmoidf_dev(float v) {
    return 1.0f / (1.0f + __expf(-v));
}

// CDNA5 async copy: 16 bytes global -> LDS, tracked by ASYNCcnt.
// LDS operand is the low 32 bits of the generic pointer (aperture rule).
__device__ __forceinline__ void async_ld16(void* lds, const void* g) {
    unsigned l = (unsigned)(size_t)lds;
    asm volatile("global_load_async_to_lds_b128 %0, %1, off"
                 :: "v"(l), "v"(g) : "memory");
}

// ---------------------------------------------------------------------------
// Problem constants: x:[16,64,128,128], weight:[64,64,3,3], se_w1:[16,64,3,3],
// se_w2:[1,16,3,3], fc_w1:[4,64], fc_w2:[576,4]
// ---------------------------------------------------------------------------
constexpr int B_ = 16, C_ = 64, H_ = 128, W_ = 128;
constexpr int HW_ = H_ * W_;

// workspace layout (bytes)
constexpr size_t OFF_MEAN = 0;                                   // 16*64*4
constexpr size_t OFF_Y    = 4096;                                // 16*576*4
constexpr size_t OFF_WP   = 40960;                               // 16*64*576*2 bf16
constexpr size_t OFF_WSE  = OFF_WP + (size_t)16 * 64 * 576 * 2;
constexpr size_t OFF_T    = OFF_WSE + (size_t)16 * 576 * 2;      // 16*128*128*16*2 bf16
constexpr size_t OFF_A    = OFF_T + (size_t)16 * 128 * 128 * 16 * 2; // 16*16384*4
constexpr size_t OFF_XH   = OFF_A + (size_t)16 * 16384 * 4;      // NHWC bf16 x: 16*16384*64*2

// ---------------------------------------------------------------------------
// K0: NCHW fp32 -> NHWC bf16 transpose of x (one row of one batch per block)
// ---------------------------------------------------------------------------
__global__ __launch_bounds__(256) void nhwc_kernel(const float* __restrict__ x,
                                                   __bf16* __restrict__ xh) {
    __shared__ __align__(16) __bf16 ls[128 * 72];  // [w][c], padded stride
    const int b = blockIdx.y, hh = blockIdx.x;
#pragma unroll
    for (int it = 0; it < 32; ++it) {              // 64c * 128w / 256
        const int idx = it * 256 + threadIdx.x;
        const int w = idx & 127, c = idx >> 7;     // lanes -> consecutive w (coalesced)
        const float v = x[(((size_t)(b * 64 + c)) * H_ + hh) * W_ + w];
        ls[w * 72 + c] = (__bf16)v;
    }
    __syncthreads();
#pragma unroll
    for (int it = 0; it < 4; ++it) {               // 1024 chunks of 8 bf16
        const int idx = it * 256 + threadIdx.x;
        const int c8 = (idx & 7) * 8, w = idx >> 3;
        const v8bf v = *(const v8bf*)&ls[w * 72 + c8];
        *(v8bf*)&xh[(((size_t)(b * H_ + hh)) * W_ + w) * 64 + c8] = v;
    }
}

// ---------------------------------------------------------------------------
// K1a: per (b,c) spatial mean of x (vectorized float4 reads)
// ---------------------------------------------------------------------------
__global__ __launch_bounds__(256) void mean_kernel(const float* __restrict__ x,
                                                   float* __restrict__ mean) {
    __shared__ float red[256];
    const int bc = blockIdx.x;                     // b*64 + c
    const float4* p = (const float4*)(x + (size_t)bc * HW_);
    float s = 0.f;
    for (int i = threadIdx.x; i < HW_ / 4; i += 256) {
        const float4 v = p[i];
        s += v.x + v.y + v.z + v.w;
    }
    red[threadIdx.x] = s;
    __syncthreads();
    for (int st = 128; st > 0; st >>= 1) {
        if ((int)threadIdx.x < st) red[threadIdx.x] += red[threadIdx.x + st];
        __syncthreads();
    }
    if (threadIdx.x == 0) mean[bc] = red[0] * (1.0f / (float)HW_);
}

// ---------------------------------------------------------------------------
// K1b: y[b,576] = sigmoid( relu(mean[b]·fc_w1^T) · fc_w2^T )
// ---------------------------------------------------------------------------
__global__ __launch_bounds__(128) void fc_kernel(const float* __restrict__ mean,
                                                 const float* __restrict__ fc_w1,
                                                 const float* __restrict__ fc_w2,
                                                 float* __restrict__ yv) {
    __shared__ float hred[4];
    const int b = blockIdx.x;
    if (threadIdx.x < 4) {
        float s = 0.f;
        for (int c = 0; c < 64; ++c) s += mean[b * 64 + c] * fc_w1[threadIdx.x * 64 + c];
        hred[threadIdx.x] = fmaxf(s, 0.f);
    }
    __syncthreads();
    for (int j = threadIdx.x; j < 576; j += blockDim.x) {
        float s = 0.f;
#pragma unroll
        for (int r = 0; r < 4; ++r) s += fc_w2[j * 4 + r] * hred[r];
        yv[b * 576 + j] = sigmoidf_dev(s);
    }
}

// ---------------------------------------------------------------------------
// K2: W'[b][o][tap][c] = weight[o][c][tap] * y[b, c*9+tap]  (bf16)
// ---------------------------------------------------------------------------
__global__ __launch_bounds__(256) void wp_kernel(const float* __restrict__ weight,
                                                 const float* __restrict__ yv,
                                                 __bf16* __restrict__ wp) {
    const int bo = blockIdx.x;                     // b*64 + o
    const int b = bo >> 6, o = bo & 63;
    for (int i = threadIdx.x; i < 576; i += 256) {
        const int kt = i >> 6, c = i & 63;
        const float wv = weight[(o * 64 + c) * 9 + kt];
        const float ys = yv[b * 576 + c * 9 + kt];
        wp[(size_t)bo * 576 + i] = (__bf16)(wv * ys);
    }
}

// K2b: Wse[o][tap][c] = se_w1[o][c][tap]  (bf16, tap-major/channel-minor)
__global__ __launch_bounds__(256) void wse_kernel(const float* __restrict__ se_w1,
                                                  __bf16* __restrict__ wse) {
    const int i = blockIdx.x * 256 + threadIdx.x;
    if (i < 16 * 576) {
        const int o = i / 576, k = i % 576;
        const int kt = k >> 6, c = k & 63;
        wse[o * 576 + k] = (__bf16)se_w1[(o * 64 + c) * 9 + kt];
    }
}

// ---------------------------------------------------------------------------
// Per-tap weight chunk stage (async to LDS): [NOUT][64] bf16, layout [n][k].
// Issues exactly NOUT*8/BD async instructions per wave.
// ---------------------------------------------------------------------------
template <int NOUT, int BD>
__device__ __forceinline__ void stage_wtap(const __bf16* __restrict__ wg, int kt,
                                           __bf16* wbuf, int tid) {
#pragma unroll
    for (int i = 0; i < (NOUT * 8) / BD; ++i) {
        const int j = i * BD + tid;
        const int o = j >> 3, c8 = (j & 7) * 8;
        async_ld16(&wbuf[o * 72 + c8], &wg[o * 576 + kt * 64 + c8]);
    }
}

// ---------------------------------------------------------------------------
// WMMA implicit-GEMM 3x3 conv, pad=1, Cin=64, x in NHWC bf16.
//  MAIN=true : NOUT=64, out[b,o,hw] = A[b,hw] * conv(x, W'[b]),  fp32 store
//  MAIN=false: NOUT=16, t[b,h,w,o]  = relu(conv(x, Wse)),        bf16 store
// Block: 64 pixels of one image row x NOUT channels. K = 9 taps * 64 ch,
// tap-major so both A and B fragments are channel-contiguous. Weight chunks
// are double-buffered in LDS and fetched with async-to-LDS one tap ahead.
// ---------------------------------------------------------------------------
template <int NOUT, bool MAIN, int BD>
__global__ __launch_bounds__(256)
void conv3x3_wmma_kernel(const __bf16* __restrict__ xh,
                         const __bf16* __restrict__ wgAll,
                         const float* __restrict__ Ag,
                         float* __restrict__ outg,
                         __bf16* __restrict__ tg) {
    constexpr int NT = (NOUT == 64) ? 2 : 1;       // D tiles per wave
    constexpr int XPAD = 72;                       // bank-conflict-free, 16B-aligned
    __shared__ __align__(16) __bf16 xs[3 * 66 * XPAD];
    __shared__ __align__(16) __bf16 wtapA[NOUT * XPAD];
    __shared__ __align__(16) __bf16 wtapB[NOUT * XPAD];

    const int tid = threadIdx.x;
    const int b = blockIdx.z, h = blockIdx.y, w0 = blockIdx.x * 64;
    const __bf16* __restrict__ wg = MAIN ? (wgAll + (size_t)b * 64 * 576) : wgAll;

    // ---- stage x tile: rows h-1..h+1, cols w0-1..w0+64 (66), 64 ch.
    // NHWC source => each chunk is 8 contiguous channels: one async b128 copy.
    for (int idx = tid; idx < 3 * 66 * 8; idx += BD) {
        const int c8   = (idx & 7) * 8;
        const int rcol = idx >> 3;
        const int col  = rcol % 66;
        const int r    = rcol / 66;
        const int hh = h + r - 1, ww = w0 + col - 1;
        __bf16* dst = &xs[(r * 66 + col) * XPAD + c8];
        if (hh >= 0 && hh < H_ && ww >= 0 && ww < W_) {
            async_ld16(dst, &xh[(((size_t)(b * H_ + hh)) * W_ + ww) * 64 + c8]);
        } else {
            v8bf z;
#pragma unroll
            for (int q = 0; q < 8; ++q) z[q] = (__bf16)0.f;
            *(v8bf*)dst = z;                        // zero halo (ds_store_b128)
        }
    }

    // preload tap 0 weights
    stage_wtap<NOUT, BD>(wg, 0, wtapA, tid);

    const int wave = tid >> 5, lane = tid & 31;
    const int mt = (NOUT == 64) ? (wave & 3) : wave;     // M tile (16 pixels)
    const int n0 = (NOUT == 64) ? ((wave >> 2) * 2) : 0; // first N tile
    const int m  = lane & 15;                            // A-frag row
    const int kh = lane >> 4;                            // K-half select

    v8f acc[NT];
#pragma unroll
    for (int t = 0; t < NT; ++t) {
#pragma unroll
        for (int r = 0; r < 8; ++r) acc[t][r] = 0.0f;
    }

    for (int kt = 0; kt < 9; ++kt) {
        const int ki = kt / 3, kj = kt % 3;
        __syncthreads();  // everyone done reading the buffer we are about to refill
        if (kt < 8) {
            // prefetch tap kt+1 into the other buffer, keep it in flight
            stage_wtap<NOUT, BD>(wg, kt + 1, (kt & 1) ? wtapA : wtapB, tid);
            if constexpr (NOUT == 64)
                asm volatile("s_wait_asynccnt 2" ::: "memory");
            else
                asm volatile("s_wait_asynccnt 1" ::: "memory");
        } else {
            asm volatile("s_wait_asynccnt 0" ::: "memory");
        }
        __syncthreads();  // tap kt data visible to all waves
        const __bf16* wcur = (kt & 1) ? wtapB : wtapA;

#pragma unroll
        for (int s = 0; s < 2; ++s) {              // 2 x K=32 steps per tap
            const int c0 = s * 32;
            // A fragment (16x32 bf16): lane<16 -> K {0..7,16..23}, lane>=16 -> {8..15,24..31}
            const int col   = mt * 16 + m + kj;
            const int abase = (ki * 66 + col) * XPAD + c0 + kh * 8;
            const v8bf alo = *(const v8bf*)&xs[abase];
            const v8bf ahi = *(const v8bf*)&xs[abase + 16];
            const v16bf af = __builtin_shufflevector(alo, ahi,
                0, 1, 2, 3, 4, 5, 6, 7, 8, 9, 10, 11, 12, 13, 14, 15);
#pragma unroll
            for (int t = 0; t < NT; ++t) {
                // B fragment (32x16 bf16): lane<16 -> K {0..7,16..23}, lane>=16 -> {8..15,24..31}
                const int n    = (n0 + t) * 16 + (lane & 15);
                const int boff = n * XPAD + c0 + (lane >> 4) * 8;
                const v8bf blo = *(const v8bf*)&wcur[boff];
                const v8bf bhi = *(const v8bf*)&wcur[boff + 16];
                const v16bf bfv = __builtin_shufflevector(blo, bhi,
                    0, 1, 2, 3, 4, 5, 6, 7, 8, 9, 10, 11, 12, 13, 14, 15);
                acc[t] = __builtin_amdgcn_wmma_f32_16x16x32_bf16(
                    false, af, false, bfv, (short)0, acc[t], false, false);
            }
        }
    }

    // ---- epilogue: D layout => n = lane&15, m = vgpr + 8*(lane>>4)
    if constexpr (MAIN) {
#pragma unroll
        for (int t = 0; t < NT; ++t) {
            const int ch = (n0 + t) * 16 + (lane & 15);
#pragma unroll
            for (int r = 0; r < 8; ++r) {
                const int ml  = mt * 16 + r + 8 * kh;
                const int pos = h * W_ + w0 + ml;
                const float aval = Ag[(size_t)b * HW_ + pos];
                outg[((size_t)(b * 64 + ch)) * HW_ + pos] = acc[t][r] * aval;
            }
        }
    } else {
        const int ch = lane & 15;
#pragma unroll
        for (int r = 0; r < 8; ++r) {
            const int ml = mt * 16 + r + 8 * kh;
            const size_t base = ((size_t)((b * H_ + h) * W_) + (w0 + ml)) * 16 + ch;
            tg[base] = (__bf16)fmaxf(acc[0][r], 0.f);
        }
    }
}

// ---------------------------------------------------------------------------
// K4: A = sigmoid(conv3x3(t, se_w2)); t is [b][h][w][16] bf16
// ---------------------------------------------------------------------------
__global__ __launch_bounds__(256) void se2_kernel(const __bf16* __restrict__ tg,
                                                  const float* __restrict__ se_w2,
                                                  float* __restrict__ Ag) {
    __shared__ float w2s[144];
    const int tid = threadIdx.x;
    if (tid < 144) w2s[tid] = se_w2[tid];
    __syncthreads();
    const int b = blockIdx.y;
    const int pix = blockIdx.x * 256 + tid;
    const int h = pix >> 7, w = pix & 127;
    float s = 0.f;
#pragma unroll
    for (int di = 0; di < 3; ++di) {
        const int hh = h + di - 1;
        if (hh < 0 || hh >= H_) continue;
#pragma unroll
        for (int dj = 0; dj < 3; ++dj) {
            const int ww = w + dj - 1;
            if (ww < 0 || ww >= W_) continue;
            const __bf16* tp = tg + ((size_t)((b * H_ + hh) * W_) + ww) * 16;
            const v8bf t0 = *(const v8bf*)tp;
            const v8bf t1 = *(const v8bf*)(tp + 8);
#pragma unroll
            for (int c = 0; c < 8; ++c) {
                s += (float)t0[c] * w2s[c * 9 + di * 3 + dj];
                s += (float)t1[c] * w2s[(c + 8) * 9 + di * 3 + dj];
            }
        }
    }
    Ag[(size_t)b * HW_ + pix] = sigmoidf_dev(s);
}

// ---------------------------------------------------------------------------
// host launcher
// ---------------------------------------------------------------------------
extern "C" void kernel_launch(void* const* d_in, const int* in_sizes, int n_in,
                              void* d_out, int out_size, void* d_ws, size_t ws_size,
                              hipStream_t stream) {
    const float* x      = (const float*)d_in[0];
    const float* weight = (const float*)d_in[1];
    const float* se_w1  = (const float*)d_in[2];
    const float* se_w2  = (const float*)d_in[3];
    const float* fc_w1  = (const float*)d_in[4];
    const float* fc_w2  = (const float*)d_in[5];
    float* out = (float*)d_out;

    char* ws = (char*)d_ws;
    float*  mean = (float*)(ws + OFF_MEAN);
    float*  yv   = (float*)(ws + OFF_Y);
    __bf16* wp   = (__bf16*)(ws + OFF_WP);
    __bf16* wse  = (__bf16*)(ws + OFF_WSE);
    __bf16* tbuf = (__bf16*)(ws + OFF_T);
    float*  Abuf = (float*)(ws + OFF_A);
    __bf16* xh   = (__bf16*)(ws + OFF_XH);

    // NHWC bf16 copy of x (feeds both WMMA convs with 16B channel chunks)
    nhwc_kernel<<<dim3(H_, B_), 256, 0, stream>>>(x, xh);

    // channel attention vector y
    mean_kernel<<<B_ * C_, 256, 0, stream>>>(x, mean);
    fc_kernel<<<B_, 128, 0, stream>>>(mean, fc_w1, fc_w2, yv);

    // effective per-batch weights (bf16, tap-major K layout)
    wp_kernel<<<B_ * C_, 256, 0, stream>>>(weight, yv, wp);
    wse_kernel<<<(16 * 576 + 255) / 256, 256, 0, stream>>>(se_w1, wse);

    // SE branch: conv1 (WMMA) -> conv2+sigmoid
    conv3x3_wmma_kernel<16, false, 128><<<dim3(2, H_, B_), 128, 0, stream>>>(
        xh, wse, nullptr, nullptr, tbuf);
    se2_kernel<<<dim3(HW_ / 256, B_), 256, 0, stream>>>(tbuf, se_w2, Abuf);

    // main per-batch conv (WMMA) fused with spatial-attention scale
    conv3x3_wmma_kernel<64, true, 256><<<dim3(2, H_, B_), 256, 0, stream>>>(
        xh, wp, Abuf, out, nullptr);
}
